// MultiHeadAttention_67757404061746
// MI455X (gfx1250) — matrix operand order repaired and checked by reading, not compiled, not run on hardware
//
#include <hip/hip_runtime.h>

// ---------------------------------------------------------------------------
// MHA for MI455X (gfx1250): bf16 WMMA GEMMs + double-buffered flash attention
// with async global->LDS staging (ASYNCcnt) where the toolchain exposes it.
//
// Workspace layout (elements; total ~75.6 MB):
//   Wt[q|k|v|o]  bf16 [K=1024][N=1024] transposed weights  (4 x 2 MB)
//   Qp           bf16 [B,H,T,D]                            (16 MB)
//   Ktp          bf16 [B,H,D,T]   (K pre-transposed)       (16 MB)
//   Vp           bf16 [B,H,T,D]                            (16 MB)
//   At           bf16 [B,T,E]                              (16 MB)
//   Mf           f32  [B,Tk] additive mask                 (32 KB)
// ---------------------------------------------------------------------------

typedef __attribute__((ext_vector_type(16))) __bf16 v16bf;
typedef __attribute__((ext_vector_type(8)))  float  v8f;
typedef __attribute__((ext_vector_type(4)))  int    v4i;

constexpr int   BATCH   = 4;
constexpr int   TQ      = 2048;
constexpr int   TK      = 2048;
constexpr int   EMB     = 1024;
constexpr int   NH      = 16;
constexpr int   DH      = 64;
constexpr int   MROWS   = BATCH * TQ;        // 8192
constexpr float SCALE_Q = 0.125f;            // 64^-0.5
constexpr float CLAMP_V = 50000.0f;
constexpr float MASK_NEG = -9.0e15f;

// ---------------------------------------------------------------------------
// Async global->LDS (gfx1250 GLOBAL_LOAD_ASYNC_TO_LDS_B128, ASYNCcnt).
// Builtin prototype (from hipcc diagnostic): (v4i as1*, v4i as3*, imm, imm).
// Guarded: falls back to VGPR round-trip copies if the builtin is absent.
// ---------------------------------------------------------------------------
#if defined(__has_builtin)
#if __has_builtin(__builtin_amdgcn_global_load_async_to_lds_b128) && \
    __has_builtin(__builtin_amdgcn_s_wait_asynccnt)
#define HAVE_ASYNC_LDS 1
#endif
#endif

typedef __attribute__((address_space(1))) v4i* as1_v4i_ptr;
typedef __attribute__((address_space(3))) v4i* as3_v4i_ptr;

__device__ __forceinline__ void copy16_async(const __bf16* g, __bf16* l) {
#if HAVE_ASYNC_LDS
    __builtin_amdgcn_global_load_async_to_lds_b128((as1_v4i_ptr)g,
                                                   (as3_v4i_ptr)l, 0, 0);
#else
    *(uint4*)l = *(const uint4*)g;
#endif
}

__device__ __forceinline__ void async_join() {
#if HAVE_ASYNC_LDS
    __builtin_amdgcn_s_wait_asynccnt(0);
#endif
}

// ---------------------------------------------------------------------------
// WMMA helpers (ISA 7.12.2 layouts, wave32)
// ---------------------------------------------------------------------------
__device__ __forceinline__ v8f wmma_bf16(v16bf a, v16bf b, v8f c) {
    return __builtin_amdgcn_wmma_f32_16x16x32_bf16(false, a, false, b,
                                                   (short)0, c, false, false);
}

// A-matrix 16x32 (MxK) fragment: lane = M; packed K pairs per ISA table.
__device__ __forceinline__ v16bf load_afrag(const __bf16* base, int rowStride,
                                            int lane, int kbase) {
    const int  m  = lane & 15;
    const bool hi = lane >= 16;
    const __bf16* row = base + m * rowStride + kbase;
    v16bf a;
#pragma unroll
    for (int v = 0; v < 8; ++v) {
        int kb = ((v < 4) ? 0 : 16) + (hi ? 8 : 0);
        int k  = kb + 2 * (v & 3);
        a[2 * v]     = row[k];
        a[2 * v + 1] = row[k + 1];
    }
    return a;
}

// B-matrix 32x16 (KxN) fragment: lane = N column; K in ISA table order.
__device__ __forceinline__ v16bf load_bfrag(const __bf16* base, int rowStride,
                                            int lane, int ncol) {
    const __bf16* col = base + ncol + (lane & 15);
    const int kh = (lane >= 16) ? 8 : 0;
    v16bf b;
#pragma unroll
    for (int e = 0; e < 16; ++e)
        b[e] = col[(((e < 8) ? 0 : 16) + kh + (e & 7)) * rowStride];
    return b;
}

// ---------------------------------------------------------------------------
// Utility kernels
// ---------------------------------------------------------------------------
// Wt[k][n] = bf16(W[n][k]) : coalesced read, scattered bf16 write (4 MB each).
__global__ void cvt_transpose_kernel(const float* __restrict__ s,
                                     __bf16* __restrict__ d) {
    int i = blockIdx.x * blockDim.x + threadIdx.x;   // over N*K
    int n = i >> 10, k = i & (EMB - 1);
    d[(size_t)k * EMB + n] = (__bf16)s[i];
}

__global__ void mask_kernel(const int* __restrict__ m, float* __restrict__ o,
                            int n) {
    int i = blockIdx.x * blockDim.x + threadIdx.x;
    if (i < n) o[i] = (m[i] == 0) ? MASK_NEG : (float)m[i];
}

// ---------------------------------------------------------------------------
// Tiled WMMA GEMM:  Out[m,n] = (sum_k A[m,k]*Wt[k,n] + bias[n]) * scale
//   MODE 0: A f32, Out bf16 head-split [B,H,T,D]      (Q, V projections)
//   MODE 2: A f32, Out bf16 head-transposed [B,H,D,T] (K projection)
//   MODE 1: A bf16, Out f32 row-major                 (final projection)
// Block: 256 thr (8 waves). Tile 128x64, wave sub-tile 32x32, K-step 32.
// B tile (and A tile for MODE 1) staged via async global->LDS b128 copies.
// ---------------------------------------------------------------------------
constexpr int AS_STR = 40;  // 80 B rows: 16 B-aligned for b128, conflict-padded
constexpr int BS_STR = 72;  // 144 B rows: 16 B-aligned for b128

template <int MODE>
__global__ __launch_bounds__(256) void gemm_kernel(
    const void* __restrict__ Ap, const __bf16* __restrict__ Wt,
    const float* __restrict__ bias, void* __restrict__ Out, float scale) {
    __shared__ __bf16 As[128 * AS_STR];   // [m][k]
    __shared__ __bf16 Bs[32 * BS_STR];    // [k][n]

    const int tid = threadIdx.x, w = tid >> 5, lane = tid & 31;
    const int wr = w >> 1, wc = w & 1;
    const int mt = blockIdx.x, nt = blockIdx.y;

    v8f acc[2][2] = {};

    for (int k0 = 0; k0 < EMB; k0 += 32) {
        // ---- stage A tile (128 rows x 32 k)
        if (MODE == 1) {
            // bf16 source: two async b128 per thread (32 B)
            const int m = tid >> 1, seg = (tid & 1) * 16;
            const __bf16* src =
                (const __bf16*)Ap + (size_t)(mt * 128 + m) * EMB + k0 + seg;
            copy16_async(src, &As[m * AS_STR + seg]);
            copy16_async(src + 8, &As[m * AS_STR + seg + 8]);
        } else {
            // f32 source: load, convert to bf16, ds_store
            const int c = (tid & 7) * 4;
#pragma unroll
            for (int p = 0; p < 4; ++p) {
                const int m = p * 32 + (tid >> 3);
                const float* src =
                    (const float*)Ap + (size_t)(mt * 128 + m) * EMB + k0 + c;
                float4 v = *(const float4*)src;
                As[m * AS_STR + c + 0] = (__bf16)v.x;
                As[m * AS_STR + c + 1] = (__bf16)v.y;
                As[m * AS_STR + c + 2] = (__bf16)v.z;
                As[m * AS_STR + c + 3] = (__bf16)v.w;
            }
        }
        // ---- stage B tile: Bs[k][n] <- Wt[k0+k][nt*64 + n], contiguous rows
        {
            const int kk = tid >> 3, seg = (tid & 7) * 8;
            copy16_async(Wt + (size_t)(k0 + kk) * EMB + nt * 64 + seg,
                         &Bs[kk * BS_STR + seg]);
        }
        // speculative prefetch of next A K-slice -> global_prefetch_b8
        if (k0 + 32 < EMB) {
            const char* pf =
                (MODE == 1)
                    ? (const char*)((const __bf16*)Ap +
                                    (size_t)(mt * 128 + (tid >> 3)) * EMB + k0 + 32)
                    : (const char*)((const float*)Ap +
                                    (size_t)(mt * 128 + (tid >> 3)) * EMB + k0 + 32);
            __builtin_prefetch(pf, 0, 1);
        }
        async_join();
        __syncthreads();

#pragma unroll
        for (int ti = 0; ti < 2; ++ti) {
            v16bf af =
                load_afrag(As + (wr * 32 + ti * 16) * AS_STR, AS_STR, lane, 0);
#pragma unroll
            for (int tj = 0; tj < 2; ++tj) {
                v16bf bf = load_bfrag(Bs, BS_STR, lane, wc * 32 + tj * 16);
                acc[ti][tj] = wmma_bf16(af, bf, acc[ti][tj]);
            }
        }
        __syncthreads();
    }

    // ---- epilogue: C layout lane=N, VGPR r -> M=r (+8 for lanes 16-31)
    const bool hi = lane >= 16;
    const int  ln = lane & 15;
#pragma unroll
    for (int ti = 0; ti < 2; ++ti)
#pragma unroll
        for (int tj = 0; tj < 2; ++tj)
#pragma unroll
            for (int r = 0; r < 8; ++r) {
                int gm = mt * 128 + wr * 32 + ti * 16 + r + (hi ? 8 : 0);
                int gn = nt * 64 + wc * 32 + tj * 16 + ln;
                float vv = (acc[ti][tj][r] + bias[gn]) * scale;
                if (MODE == 1) {
                    ((float*)Out)[(size_t)gm * EMB + gn] = vv;
                } else {
                    int bidx = gm >> 11, trow = gm & (TQ - 1);
                    int hh = gn >> 6, dd = gn & 63;
                    if (MODE == 0) {
                        ((__bf16*)Out)[(((size_t)bidx * NH + hh) * TQ + trow) *
                                           DH + dd] = (__bf16)vv;
                    } else {  // MODE 2: [B,H,D,T] for K
                        ((__bf16*)Out)[(((size_t)bidx * NH + hh) * DH + dd) *
                                           TK + trow] = (__bf16)vv;
                    }
                }
            }
}

// ---------------------------------------------------------------------------
// Flash attention per (b, h, 64-row q tile). 128 thr = 4 waves, each wave
// owns 16 q rows. Online softmax over 32 chunks of 64 keys; K/V chunks are
// double-buffered in LDS with async staging overlapping the WMMA work.
// ---------------------------------------------------------------------------
__global__ __launch_bounds__(128) void attn_kernel(
    const __bf16* __restrict__ Qp, const __bf16* __restrict__ Ktp,
    const __bf16* __restrict__ Vp, const float* __restrict__ maskf,
    __bf16* __restrict__ Ob) {
    __shared__ __bf16 Qs[64 * 64];         // [qrow][d]
    __shared__ __bf16 Ks[2][64 * 64];      // [d][key]  (B operand of Q.K^T)
    __shared__ __bf16 Vs[2][64 * 64];      // [key][d]  (B operand of P.V)
    __shared__ __bf16 Ps[4][16 * 64];      // per-wave P tile [qrow][key]

    const int tid = threadIdx.x, w = tid >> 5, lane = tid & 31;
    const int qt = blockIdx.x, h = blockIdx.y, bb = blockIdx.z;
    const size_t bh = (size_t)bb * NH + h;
    const bool hi = lane >= 16;
    const int  ln = lane & 15;

    // contiguous-row chunk staging: thread covers 64 B (4 x b128)
    const int srow = tid >> 1, sseg = (tid & 1) * 32;
    auto stage_kv = [&](int jc, int buf) {
        const __bf16* sk = Ktp + (bh * DH + srow) * TK + (size_t)jc * 64 + sseg;
        const __bf16* sv = Vp + (bh * TK + (size_t)jc * 64 + srow) * DH + sseg;
        __bf16* dk = &Ks[buf][srow * 64 + sseg];
        __bf16* dv = &Vs[buf][srow * 64 + sseg];
#pragma unroll
        for (int i = 0; i < 4; ++i) {
            copy16_async(sk + i * 8, dk + i * 8);
            copy16_async(sv + i * 8, dv + i * 8);
        }
    };

    // ---- stage Q tile + first K/V chunk
    {
        const __bf16* sq = Qp + (bh * TQ + (size_t)qt * 64 + srow) * DH + sseg;
        __bf16* dq = &Qs[srow * 64 + sseg];
#pragma unroll
        for (int i = 0; i < 4; ++i) copy16_async(sq + i * 8, dq + i * 8);
    }
    stage_kv(0, 0);
    async_join();
    __syncthreads();

    v16bf qfrag[2];
    qfrag[0] = load_afrag(Qs + w * 16 * 64, 64, lane, 0);
    qfrag[1] = load_afrag(Qs + w * 16 * 64, 64, lane, 32);

    v8f oacc[4] = {};
    float mrow[8], lrow[8];
#pragma unroll
    for (int r = 0; r < 8; ++r) { mrow[r] = -3.0e38f; lrow[r] = 0.0f; }

    constexpr int NC = TK / 64;
    for (int jc = 0; jc < NC; ++jc) {
        const int cur = jc & 1;
        if (jc + 1 < NC) stage_kv(jc + 1, cur ^ 1);  // overlap with compute

        // ---- S = Q.K^T  (4 N-tiles x 2 K-steps = 8 WMMA)
        v8f s[4];
#pragma unroll
        for (int nt = 0; nt < 4; ++nt) {
            v8f z = {};
            s[nt] = z;
#pragma unroll
            for (int ks = 0; ks < 2; ++ks) {
                v16bf bf =
                    load_bfrag(Ks[cur] + ks * 32 * 64, 64, lane, nt * 16);
                s[nt] = wmma_bf16(qfrag[ks], bf, s[nt]);
            }
        }

        // ---- clamp + additive mask + chunk row-max
        float am[4];
#pragma unroll
        for (int nt = 0; nt < 4; ++nt)
            am[nt] = maskf[bb * TK + jc * 64 + nt * 16 + ln];
        float cmax[8];
#pragma unroll
        for (int r = 0; r < 8; ++r) cmax[r] = -3.0e38f;
#pragma unroll
        for (int nt = 0; nt < 4; ++nt)
#pragma unroll
            for (int r = 0; r < 8; ++r) {
                float x = fminf(fmaxf(s[nt][r], -CLAMP_V), CLAMP_V) + am[nt];
                s[nt][r] = x;
                cmax[r] = fmaxf(cmax[r], x);
            }
#pragma unroll
        for (int off = 1; off < 16; off <<= 1)
#pragma unroll
            for (int r = 0; r < 8; ++r)
                cmax[r] = fmaxf(cmax[r], __shfl_xor(cmax[r], off, 32));

        // ---- online softmax update
        float alpha[8];
#pragma unroll
        for (int r = 0; r < 8; ++r) {
            float mn = fmaxf(mrow[r], cmax[r]);
            alpha[r] = __expf(mrow[r] - mn);
            mrow[r] = mn;
        }
        float csum[8];
#pragma unroll
        for (int r = 0; r < 8; ++r) csum[r] = 0.0f;
#pragma unroll
        for (int nt = 0; nt < 4; ++nt)
#pragma unroll
            for (int r = 0; r < 8; ++r) {
                float p = __expf(s[nt][r] - mrow[r]);
                csum[r] += p;
                Ps[w][(r + (hi ? 8 : 0)) * 64 + nt * 16 + ln] = (__bf16)p;
            }
#pragma unroll
        for (int off = 1; off < 16; off <<= 1)
#pragma unroll
            for (int r = 0; r < 8; ++r)
                csum[r] += __shfl_xor(csum[r], off, 32);
#pragma unroll
        for (int r = 0; r < 8; ++r) lrow[r] = lrow[r] * alpha[r] + csum[r];
#pragma unroll
        for (int nt = 0; nt < 4; ++nt)
#pragma unroll
            for (int r = 0; r < 8; ++r) oacc[nt][r] *= alpha[r];

        // ---- O += P.V  (same-wave LDS RAW: in-order; compiler waits dscnt)
        v16bf pfr0 = load_afrag(Ps[w], 64, lane, 0);
        v16bf pfr1 = load_afrag(Ps[w], 64, lane, 32);
#pragma unroll
        for (int nt = 0; nt < 4; ++nt) {
            v16bf b0 = load_bfrag(Vs[cur], 64, lane, nt * 16);
            oacc[nt] = wmma_bf16(pfr0, b0, oacc[nt]);
            v16bf b1 = load_bfrag(Vs[cur] + 32 * 64, 64, lane, nt * 16);
            oacc[nt] = wmma_bf16(pfr1, b1, oacc[nt]);
        }
        async_join();      // next chunk resident before anyone proceeds
        __syncthreads();
    }

    // ---- normalize and write [B,T,E] bf16
#pragma unroll
    for (int nt = 0; nt < 4; ++nt)
#pragma unroll
        for (int r = 0; r < 8; ++r) {
            int row = qt * 64 + w * 16 + r + (hi ? 8 : 0);
            int d = nt * 16 + ln;
            float val = oacc[nt][r] / lrow[r];
            Ob[((size_t)bb * TQ + row) * EMB + h * DH + d] = (__bf16)val;
        }
}

// ---------------------------------------------------------------------------
extern "C" void kernel_launch(void* const* d_in, const int* in_sizes, int n_in,
                              void* d_out, int out_size, void* d_ws,
                              size_t ws_size, hipStream_t stream) {
    (void)in_sizes; (void)n_in; (void)out_size; (void)ws_size;
    const float* q  = (const float*)d_in[0];
    const float* k  = (const float*)d_in[1];
    const float* v  = (const float*)d_in[2];
    const int*   am = (const int*)d_in[3];
    const float* Wq = (const float*)d_in[4];
    const float* bq = (const float*)d_in[5];
    const float* Wk = (const float*)d_in[6];
    const float* bk = (const float*)d_in[7];
    const float* Wv = (const float*)d_in[8];
    const float* bv = (const float*)d_in[9];
    const float* Wo = (const float*)d_in[10];
    const float* bo = (const float*)d_in[11];
    float* out = (float*)d_out;

    constexpr size_t WB  = (size_t)EMB * EMB;            // weight elems
    constexpr size_t QKV = (size_t)BATCH * NH * TQ * DH; // per-tensor elems
    __bf16* Wqt = (__bf16*)d_ws;
    __bf16* Wkt = Wqt + WB;
    __bf16* Wvt = Wkt + WB;
    __bf16* Wot = Wvt + WB;
    __bf16* Qp  = Wot + WB;
    __bf16* Ktp = Qp + QKV;
    __bf16* Vp  = Ktp + QKV;
    __bf16* At  = Vp + QKV;
    float*  Mf  = (float*)(At + (size_t)BATCH * TQ * EMB);

    // 1) weights -> transposed bf16, additive mask -> f32
    const int cvtGrid = (int)(WB / 256);
    cvt_transpose_kernel<<<cvtGrid, 256, 0, stream>>>(Wq, Wqt);
    cvt_transpose_kernel<<<cvtGrid, 256, 0, stream>>>(Wk, Wkt);
    cvt_transpose_kernel<<<cvtGrid, 256, 0, stream>>>(Wv, Wvt);
    cvt_transpose_kernel<<<cvtGrid, 256, 0, stream>>>(Wo, Wot);
    mask_kernel<<<(BATCH * TK) / 256, 256, 0, stream>>>(am, Mf, BATCH * TK);

    // 2) projections (WMMA); q path carries SCALE, K writes [B,H,D,T]
    dim3 ggrid(MROWS / 128, EMB / 64);
    gemm_kernel<0><<<ggrid, 256, 0, stream>>>(q, Wqt, bq, Qp, SCALE_Q);
    gemm_kernel<2><<<ggrid, 256, 0, stream>>>(k, Wkt, bk, Ktp, 1.0f);
    gemm_kernel<0><<<ggrid, 256, 0, stream>>>(v, Wvt, bv, Vp, 1.0f);

    // 3) flash attention (double-buffered async K/V staging)
    attn_kernel<<<dim3(TQ / 64, NH, BATCH), 128, 0, stream>>>(Qp, Ktp, Vp, Mf,
                                                              At);

    // 4) output projection (WMMA), f32 out
    gemm_kernel<1><<<ggrid, 256, 0, stream>>>(At, Wot, bo, out, 1.0f);
}
